// CustomMultiHeadAttention_41128606826521
// MI455X (gfx1250) — matrix operand order repaired and checked by reading, not compiled, not run on hardware
//
#include <hip/hip_runtime.h>

// ---------------------------------------------------------------------------
// CDNA5 (gfx1250) causal multi-head attention forward.
// All matmuls via v_wmma_f32_16x16x32_bf16 (f32 accumulate).
// ---------------------------------------------------------------------------

typedef __attribute__((ext_vector_type(16))) __bf16 v16bf;
typedef __attribute__((ext_vector_type(8)))  __bf16 v8bf;
typedef __attribute__((ext_vector_type(8)))  float  v8f;

#define D_MODEL  1024
#define N_HEADS  16
#define HEAD_DIM 64
#define SEQ      2048
#define BATCH    2
#define M_TOTAL  (BATCH * SEQ)      // 4096
#define N_QKV    (3 * D_MODEL)      // 3072

// round-to-nearest-even f32 -> bf16 bits (avoids relying on scalar __bf16 ops)
__device__ __forceinline__ unsigned short f2bf_bits(float f) {
  union { float f; unsigned u; } v; v.f = f;
  unsigned r = v.u + 0x7FFFu + ((v.u >> 16) & 1u);
  return (unsigned short)(r >> 16);
}

// ---- WMMA fragment loaders (layouts per CDNA5 ISA 7.12.2) ------------------
// A fragment: 16x32 bf16. lane L holds row (L&15); element e:
//   K = (e/8)*16 + (L>>4)*8 + (e%8)  -> two contiguous 16B chunks.
__device__ __forceinline__ v16bf load_A_frag(const unsigned short* src,
                                             int row0, int k0, int ld) {
  const int lane = threadIdx.x & 31;
  const unsigned short* p =
      src + (size_t)(row0 + (lane & 15)) * ld + k0 + (lane >> 4) * 8;
  union { v16bf v; v8bf h[2]; } u;
  u.h[0] = *(const v8bf*)(p);
  u.h[1] = *(const v8bf*)(p + 16);
  return u.v;
}

// B fragment: 32x16 bf16, B[K][N] = src[(col0+N)*ld + k0 + K]
// lane L holds col (L&15); element e: K = (e/8)*16 + (L>>4)*8 + (e%8) -> two 16B chunks.
__device__ __forceinline__ v16bf load_B_frag(const unsigned short* src,
                                             int col0, int k0, int ld) {
  const int lane = threadIdx.x & 31;
  const unsigned short* p =
      src + (size_t)(col0 + (lane & 15)) * ld + k0 + (lane >> 4) * 8;
  union { v16bf v; v8bf h[2]; } u;
  u.h[0] = *(const v8bf*)(p);
  u.h[1] = *(const v8bf*)(p + 16);
  return u.v;
}

__device__ __forceinline__ v8f wmma_bf16(v16bf a, v16bf b, v8f c) {
  return __builtin_amdgcn_wmma_f32_16x16x32_bf16(
      /*neg_a=*/false, a, /*neg_b=*/false, b,
      /*c_mod=*/(short)0, c, /*reuse_a=*/false, /*reuse_b=*/false);
}

// ---------------------------------------------------------------------------
__global__ void convert_kernel(const float* __restrict__ src,
                               unsigned short* __restrict__ dst, int n) {
  int i = blockIdx.x * blockDim.x + threadIdx.x;
  if (i < n) dst[i] = f2bf_bits(src[i]);
}

// ---------------------------------------------------------------------------
// QKV projection: [4096,1024] x [1024,3072] (+bias), scatter to per-head
// Q (scaled by 1/8), K row-major [bh][t][64]; V transposed [bh][64][T].
// One wave computes a 64x64 tile.
__global__ __launch_bounds__(32) void qkv_gemm_kernel(
    const unsigned short* __restrict__ xb,
    const unsigned short* __restrict__ wb,   // qkv_w bf16 [3072][1024]
    const float* __restrict__ bias,          // [3072]
    unsigned short* __restrict__ Qs,
    unsigned short* __restrict__ Ks,
    unsigned short* __restrict__ Vt) {
  const int n0 = blockIdx.x * 64;
  const int m0 = blockIdx.y * 64;
  const v8f zero = {0.f, 0.f, 0.f, 0.f, 0.f, 0.f, 0.f, 0.f};
  v8f acc[4][4];
#pragma unroll
  for (int i = 0; i < 4; ++i)
#pragma unroll
    for (int j = 0; j < 4; ++j) acc[i][j] = zero;

  for (int k0 = 0; k0 < D_MODEL; k0 += 32) {
    v16bf a[4], b[4];
#pragma unroll
    for (int i = 0; i < 4; ++i) a[i] = load_A_frag(xb, m0 + i * 16, k0, D_MODEL);
#pragma unroll
    for (int j = 0; j < 4; ++j) b[j] = load_B_frag(wb, n0 + j * 16, k0, D_MODEL);
#pragma unroll
    for (int i = 0; i < 4; ++i)
#pragma unroll
      for (int j = 0; j < 4; ++j) acc[i][j] = wmma_bf16(a[i], b[j], acc[i][j]);
  }

  const int lane = threadIdx.x & 31;
  const int col = lane & 15;
  const int rbase = (lane >> 4) * 8;
#pragma unroll
  for (int i = 0; i < 4; ++i)
#pragma unroll
    for (int j = 0; j < 4; ++j)
#pragma unroll
      for (int r = 0; r < 8; ++r) {
        int m = m0 + i * 16 + rbase + r;
        int n = n0 + j * 16 + col;
        float v = acc[i][j][r] + bias[n];
        int bb = m >> 11, t = m & (SEQ - 1);
        int which = n >> 10;
        int hn = n & (D_MODEL - 1);
        int h = hn >> 6, d = hn & 63;
        size_t bh = (size_t)(bb * N_HEADS + h);
        if (which == 0) {
          Qs[(bh * SEQ + t) * HEAD_DIM + d] = f2bf_bits(v * 0.125f);  // 1/sqrt(64)
        } else if (which == 1) {
          Ks[(bh * SEQ + t) * HEAD_DIM + d] = f2bf_bits(v);
        } else {
          Vt[(bh * HEAD_DIM + d) * SEQ + t] = f2bf_bits(v);
        }
      }
}

// ---------------------------------------------------------------------------
// Flash attention: one wave per (bh, 64-row q-tile). Online softmax in f32.
__global__ __launch_bounds__(32) void attn_kernel(
    const unsigned short* __restrict__ Qs,
    const unsigned short* __restrict__ Ks,
    const unsigned short* __restrict__ Vt,
    unsigned short* __restrict__ Os) {
  const int qt = blockIdx.x;   // q tile index (SEQ/64 = 32)
  const int bh = blockIdx.y;   // b*16 + h    (32)
  const unsigned short* Qh = Qs + (size_t)bh * SEQ * HEAD_DIM;
  const unsigned short* Kh = Ks + (size_t)bh * SEQ * HEAD_DIM;
  const unsigned short* Vh = Vt + (size_t)bh * HEAD_DIM * SEQ;

  __shared__ __align__(32) unsigned short Pl[64 * 64];  // P tile staging (8KB)

  const int lane = threadIdx.x & 31;
  const int col = lane & 15;
  const int rbase = (lane >> 4) * 8;
  const int q0 = qt * 64;
  const v8f zero = {0.f, 0.f, 0.f, 0.f, 0.f, 0.f, 0.f, 0.f};

  v8f o[4][4];
  float mst[4][8], lst[4][8];
#pragma unroll
  for (int i = 0; i < 4; ++i) {
#pragma unroll
    for (int j = 0; j < 4; ++j) o[i][j] = zero;
#pragma unroll
    for (int r = 0; r < 8; ++r) { mst[i][r] = -1e30f; lst[i][r] = 0.0f; }
  }

  // Q fragments are invariant over the kt loop: hoist (2 k-steps x 4 subtiles)
  v16bf qa[2][4];
#pragma unroll
  for (int kk = 0; kk < 2; ++kk)
#pragma unroll
    for (int i = 0; i < 4; ++i)
      qa[kk][i] = load_A_frag(Qh, q0 + i * 16, kk * 32, HEAD_DIM);

  for (int kt = 0; kt <= qt; ++kt) {
    const int kbase = kt * 64;

    // S = Q * K^T  (64x64)
    v8f s[4][4];
#pragma unroll
    for (int i = 0; i < 4; ++i)
#pragma unroll
      for (int j = 0; j < 4; ++j) s[i][j] = zero;
#pragma unroll
    for (int kk = 0; kk < 2; ++kk) {
      v16bf b[4];
#pragma unroll
      for (int j = 0; j < 4; ++j)
        b[j] = load_B_frag(Kh, kbase + j * 16, kk * 32, HEAD_DIM);
#pragma unroll
      for (int i = 0; i < 4; ++i)
#pragma unroll
        for (int j = 0; j < 4; ++j) s[i][j] = wmma_bf16(qa[kk][i], b[j], s[i][j]);
    }

    // causal mask on the diagonal tile
    if (kt == qt) {
#pragma unroll
      for (int i = 0; i < 4; ++i)
#pragma unroll
        for (int j = 0; j < 4; ++j)
#pragma unroll
          for (int r = 0; r < 8; ++r) {
            int qrow = i * 16 + rbase + r;
            int kcol = j * 16 + col;
            if (kcol > qrow) s[i][j][r] = -1e30f;
          }
    }

    // online softmax; rows live across 16 lanes of a half-wave
#pragma unroll
    for (int i = 0; i < 4; ++i)
#pragma unroll
      for (int r = 0; r < 8; ++r) {
        float v = s[i][0][r];
        v = fmaxf(v, s[i][1][r]);
        v = fmaxf(v, s[i][2][r]);
        v = fmaxf(v, s[i][3][r]);
        v = fmaxf(v, __shfl_xor(v, 1, 32));
        v = fmaxf(v, __shfl_xor(v, 2, 32));
        v = fmaxf(v, __shfl_xor(v, 4, 32));
        v = fmaxf(v, __shfl_xor(v, 8, 32));
        float mnew = fmaxf(mst[i][r], v);
        float corr = __expf(mst[i][r] - mnew);
        mst[i][r] = mnew;
        float rs = 0.0f;
#pragma unroll
        for (int j = 0; j < 4; ++j) {
          float p = __expf(s[i][j][r] - mnew);
          s[i][j][r] = p;
          rs += p;
          o[i][j][r] *= corr;
        }
        rs += __shfl_xor(rs, 1, 32);
        rs += __shfl_xor(rs, 2, 32);
        rs += __shfl_xor(rs, 4, 32);
        rs += __shfl_xor(rs, 8, 32);
        lst[i][r] = lst[i][r] * corr + rs;
      }

    __syncthreads();
    // C-layout -> LDS (row-major bf16 P tile) for A-layout reload
#pragma unroll
    for (int i = 0; i < 4; ++i)
#pragma unroll
      for (int j = 0; j < 4; ++j)
#pragma unroll
        for (int r = 0; r < 8; ++r)
          Pl[(i * 16 + rbase + r) * 64 + j * 16 + col] = f2bf_bits(s[i][j][r]);
    __syncthreads();

    // O += P * V   (V stored transposed: Vh[d][t], contiguous B-fragments)
#pragma unroll
    for (int kk = 0; kk < 2; ++kk) {
      v16bf pa[4], bv[4];
#pragma unroll
      for (int i = 0; i < 4; ++i) pa[i] = load_A_frag(Pl, i * 16, kk * 32, 64);
#pragma unroll
      for (int j = 0; j < 4; ++j)
        bv[j] = load_B_frag(Vh, j * 16, kbase + kk * 32, SEQ);
#pragma unroll
      for (int i = 0; i < 4; ++i)
#pragma unroll
        for (int j = 0; j < 4; ++j) o[i][j] = wmma_bf16(pa[i], bv[j], o[i][j]);
    }
    __syncthreads();
  }

  // normalize and store O [bh][t][64] bf16
#pragma unroll
  for (int i = 0; i < 4; ++i)
#pragma unroll
    for (int j = 0; j < 4; ++j)
#pragma unroll
      for (int r = 0; r < 8; ++r) {
        int t = q0 + i * 16 + rbase + r;
        int d = j * 16 + col;
        float v = o[i][j][r] / lst[i][r];
        Os[((size_t)bh * SEQ + t) * HEAD_DIM + d] = f2bf_bits(v);
      }
}

// ---------------------------------------------------------------------------
// Output projection: out[m,n] = sum_k O[b,h,t,d] * out_w[n,k] + out_b[n]
// with m=(b,t), k=h*64+d.  One wave per 64x64 tile.  f32 output.
__global__ __launch_bounds__(32) void out_gemm_kernel(
    const unsigned short* __restrict__ Os,
    const unsigned short* __restrict__ wb,   // out_w bf16 [1024][1024]
    const float* __restrict__ bias,          // [1024]
    float* __restrict__ out) {
  const int n0 = blockIdx.x * 64;
  const int m0 = blockIdx.y * 64;
  const int lane = threadIdx.x & 31;
  const v8f zero = {0.f, 0.f, 0.f, 0.f, 0.f, 0.f, 0.f, 0.f};
  v8f acc[4][4];
#pragma unroll
  for (int i = 0; i < 4; ++i)
#pragma unroll
    for (int j = 0; j < 4; ++j) acc[i][j] = zero;

  for (int k0 = 0; k0 < D_MODEL; k0 += 32) {
    const int h = k0 >> 6;          // 32-wide k-chunk never crosses a head
    const int dbase = k0 & 63;
    v16bf a[4], b[4];
#pragma unroll
    for (int i = 0; i < 4; ++i) {
      int m = m0 + i * 16 + (lane & 15);
      int bb = m >> 11, t = m & (SEQ - 1);
      const unsigned short* p =
          Os + ((size_t)(bb * N_HEADS + h) * SEQ + t) * HEAD_DIM + dbase +
          (lane >> 4) * 8;
      union { v16bf v; v8bf hh[2]; } u;
      u.hh[0] = *(const v8bf*)(p);
      u.hh[1] = *(const v8bf*)(p + 16);
      a[i] = u.v;
    }
#pragma unroll
    for (int j = 0; j < 4; ++j) b[j] = load_B_frag(wb, n0 + j * 16, k0, D_MODEL);
#pragma unroll
    for (int i = 0; i < 4; ++i)
#pragma unroll
      for (int j = 0; j < 4; ++j) acc[i][j] = wmma_bf16(a[i], b[j], acc[i][j]);
  }

  const int col = lane & 15;
  const int rbase = (lane >> 4) * 8;
#pragma unroll
  for (int i = 0; i < 4; ++i)
#pragma unroll
    for (int j = 0; j < 4; ++j)
#pragma unroll
      for (int r = 0; r < 8; ++r) {
        int m = m0 + i * 16 + rbase + r;
        int n = n0 + j * 16 + col;
        out[(size_t)m * D_MODEL + n] = acc[i][j][r] + bias[n];
      }
}

// ---------------------------------------------------------------------------
extern "C" void kernel_launch(void* const* d_in, const int* in_sizes, int n_in,
                              void* d_out, int out_size, void* d_ws,
                              size_t ws_size, hipStream_t stream) {
  (void)in_sizes; (void)n_in; (void)out_size; (void)ws_size;
  const float* x     = (const float*)d_in[0];
  const float* qkv_w = (const float*)d_in[1];
  const float* qkv_b = (const float*)d_in[2];
  const float* out_w = (const float*)d_in[3];
  const float* out_b = (const float*)d_in[4];
  float* out = (float*)d_out;

  // workspace layout (48 MB total)
  char* ws = (char*)d_ws;
  unsigned short* xb  = (unsigned short*)(ws + 0);                        // 8 MB
  unsigned short* qwb = (unsigned short*)(ws + (size_t)8  * 1024 * 1024); // 6 MB
  unsigned short* owb = (unsigned short*)(ws + (size_t)14 * 1024 * 1024); // 2 MB
  unsigned short* Qs  = (unsigned short*)(ws + (size_t)16 * 1024 * 1024); // 8 MB
  unsigned short* Ks  = (unsigned short*)(ws + (size_t)24 * 1024 * 1024); // 8 MB
  unsigned short* Vt  = (unsigned short*)(ws + (size_t)32 * 1024 * 1024); // 8 MB
  unsigned short* Os  = (unsigned short*)(ws + (size_t)40 * 1024 * 1024); // 8 MB

  convert_kernel<<<(M_TOTAL * D_MODEL + 255) / 256, 256, 0, stream>>>(
      x, xb, M_TOTAL * D_MODEL);
  convert_kernel<<<(N_QKV * D_MODEL + 255) / 256, 256, 0, stream>>>(
      qkv_w, qwb, N_QKV * D_MODEL);
  convert_kernel<<<(D_MODEL * D_MODEL + 255) / 256, 256, 0, stream>>>(
      out_w, owb, D_MODEL * D_MODEL);

  qkv_gemm_kernel<<<dim3(N_QKV / 64, M_TOTAL / 64), 32, 0, stream>>>(
      xb, qwb, qkv_b, Qs, Ks, Vt);

  attn_kernel<<<dim3(SEQ / 64, BATCH * N_HEADS), 32, 0, stream>>>(Qs, Ks, Vt, Os);

  out_gemm_kernel<<<dim3(D_MODEL / 64, M_TOTAL / 64), 32, 0, stream>>>(
      Os, owb, out_b, out);
}
